// RNNWithAttention_33930241639042
// MI455X (gfx1250) — matrix operand order repaired and checked by reading, not compiled, not run on hardware
//
#include <hip/hip_runtime.h>
#include <cstdint>
#include <cstddef>

// ---------------------------------------------------------------------------
// RNN-with-attention for MI455X (gfx1250).
//   B=64, L=512, D=H=256, OUT=2.
// Strategy:
//   0) pre-convert x -> bf16 in [l*B+b, k] row order (GEMM A layout),
//      Wh/Wx -> bf16 (B operands; B col n == W row n, contiguous loads)
//   1) P[l,b,:] = x[b,l,:] @ Wx^T + bx + bh   -- big bf16 WMMA GEMM
//   2) fwd/bwd LSTM scans: 8 persistent WGs (4 batch groups x 2 dirs),
//      h tile in LDS (bf16 A-matrix), c-state in registers, WMMA per step;
//      P tile double-buffered in LDS via global_load_async_to_lds_b128
//   3) enc = 0.5*(hfwd+hbwd)
//   4) attention scan: 64 cooperating WGs + global barrier; energies /
//      softmax / context stream enc from L2, cell = 1 WMMA tile per WG
//   5) out = listH @ Wl^T + bl, masked by slen
// ---------------------------------------------------------------------------

#define Bsz 64
#define Lsz 512
#define Dsz 256
#define Hsz 256

typedef __attribute__((ext_vector_type(16))) __bf16 v16bf;
typedef __attribute__((ext_vector_type(8)))  float  v8f;
typedef __attribute__((ext_vector_type(4)))  unsigned int u32x4;

union V16BF {
  v16bf v;
  u32x4 q[2];
  unsigned short h[16];
};

// round-to-nearest-even f32 -> bf16 (bit trick; NaN not special-cased)
__device__ __forceinline__ unsigned short f2bf(float f) {
  unsigned int u = __float_as_uint(f);
  unsigned int r = (u + 0x7FFFu + ((u >> 16) & 1u)) >> 16;
  return (unsigned short)r;
}

__device__ __forceinline__ float sigm(float x) {
  return 1.0f / (1.0f + __expf(-x));
}

// ---- CDNA5 async global->LDS copy (ISA §15.18, tracked by ASYNCcnt) -------
// vdst = wave-relative LDS byte offset (low 32 bits of generic LDS pointer),
// vaddr = 64-bit global address.
__device__ __forceinline__ void async_copy_b128(const void* gaddr, void* lds) {
  unsigned int loff = (unsigned int)(uintptr_t)lds;
  asm volatile("global_load_async_to_lds_b128 %0, %1, off"
               :: "v"(loff), "v"(gaddr) : "memory");
}
__device__ __forceinline__ void wait_asynccnt0() {
  asm volatile("s_wait_asynccnt 0x0" ::: "memory");
}

// ---------------------------------------------------------------------------
// float -> bf16 conversion of weight matrices
// ---------------------------------------------------------------------------
__global__ void cvt_bf16_kernel(const float* __restrict__ src,
                                unsigned short* __restrict__ dst, int n) {
  int i = blockIdx.x * blockDim.x + threadIdx.x;
  if (i < n) dst[i] = f2bf(src[i]);
}

// x[b,l,k] (f32) -> Xbf[l*B+b, k] (bf16): A-operand layout for xproj GEMM
__global__ void cvt_x_kernel(const float* __restrict__ x,
                             unsigned short* __restrict__ Xbf) {
  const size_t o = (size_t)blockIdx.x * blockDim.x + threadIdx.x;
  const int k = (int)(o & 255);
  const size_t row = o >> 8;           // l*64 + b
  const int b = (int)(row & 63);
  const int l = (int)(row >> 6);
  Xbf[o] = f2bf(x[(((size_t)b << 9) + l) * Dsz + k]);
}

// ---------------------------------------------------------------------------
// P[l,b,n] = sum_k x[b,l,k]*Wx[n,k] + bx[n] + bh[n]
// One 16x16 output tile per wave; K loop of 8 x wmma_f32_16x16x32_bf16.
// A from pre-converted Xbf (contiguous 16B chunks per lane per ISA layout),
// B from pre-converted Wxbf.
// ---------------------------------------------------------------------------
__global__ void xproj_kernel(const unsigned short* __restrict__ Xbf,
                             const unsigned short* __restrict__ Wxbf,
                             const float* __restrict__ bx,
                             const float* __restrict__ bh,
                             float* __restrict__ P) {
  const int tid  = threadIdx.x;
  const int wave = tid >> 5;
  const int lane = tid & 31;
  const int col  = lane & 15;
  const int hi   = lane >> 4;

  const int W  = blockIdx.x * 8 + wave;   // global tile id, 32768 tiles
  const int mt = W >> 4;                  // M tile (rows of L*B)
  const int nt = W & 15;                  // N tile (hidden cols)

  const float bias = bx[nt * 16 + col] + bh[nt * 16 + col];
  v8f acc;
#pragma unroll
  for (int r = 0; r < 8; ++r) acc[r] = bias;

  const unsigned short* xr = Xbf + (size_t)(mt * 16 + col) * Dsz;

#pragma unroll
  for (int kc = 0; kc < 8; ++kc) {
    V16BF am, bm;
    const int k0 = kc * 32 + hi * 8;
    am.q[0] = *(const u32x4*)(xr + k0);
    am.q[1] = *(const u32x4*)(xr + k0 + 16);

    const unsigned short* wr =
        Wxbf + (size_t)(nt * 16 + col) * Dsz + kc * 32 + hi * 8;
    bm.q[0] = *(const u32x4*)(wr);
    bm.q[1] = *(const u32x4*)(wr + 16);

    acc = __builtin_amdgcn_wmma_f32_16x16x32_bf16(
        false, am.v, false, bm.v, (short)0, acc, false, false);
  }

#pragma unroll
  for (int r = 0; r < 8; ++r) {
    const size_t row = (size_t)(mt * 16 + 8 * hi + r);
    P[row * Hsz + nt * 16 + col] = acc[r];
  }
}

// ---------------------------------------------------------------------------
// Forward / backward LSTM scans.
// grid = 8 WGs: wg&1 = direction, wg>>1 = batch group (16 rows).
// block = 256 (8 waves); wave owns N-tiles {2w, 2w+1}; c-state in registers.
// h tile (16 x 256 bf16) lives in LDS and is the WMMA A-matrix each step.
// The 16KB contiguous P tile for the NEXT step is async-staged into a
// double-buffered LDS region while the current step's WMMAs run.
// ---------------------------------------------------------------------------
__global__ void scan_kernel(const float* __restrict__ P,
                            const unsigned short* __restrict__ Whbf,
                            float* __restrict__ hfwd,
                            float* __restrict__ hbwd,
                            float* __restrict__ hstate,
                            float* __restrict__ cstate) {
  const int wg  = blockIdx.x;
  const int dir = wg & 1;
  const int bg  = wg >> 1;
  const int tid = threadIdx.x;
  const int wave = tid >> 5;
  const int lane = tid & 31;
  const int col  = lane & 15;
  const int hi   = lane >> 4;
  const int nt0  = wave * 2;
  const int nt1  = wave * 2 + 1;

  __shared__ unsigned short hlds[16 * Hsz];   // 8 KB h tile (bf16)
  __shared__ float ptile[2][16 * Hsz];        // 32 KB P tile double buffer

  for (int i = tid; i < 16 * Hsz; i += 256) hlds[i] = 0;

  float cs0[8], cs1[8], hl0[8], hl1[8];
#pragma unroll
  for (int r = 0; r < 8; ++r) { cs0[r] = 0.f; cs1[r] = 0.f; hl0[r] = 0.f; hl1[r] = 0.f; }

  float* hbuf = dir ? hbwd : hfwd;

  // stage P tile for step 0: rows [t0*64 + bg*16, +16) are contiguous (16KB)
  {
    const int t0 = dir ? (Lsz - 1) : 0;
    const float* Pt = P + ((size_t)t0 * Bsz + bg * 16) * Hsz;
#pragma unroll
    for (int j = 0; j < 4; ++j) {
      const int idx = tid + j * 256;          // b128 chunk index, 0..1023
      async_copy_b128(Pt + idx * 4, &ptile[0][idx * 4]);
    }
  }

  int buf = 0;
  for (int s = 0; s < Lsz; ++s) {
    wait_asynccnt0();      // this step's P tile landed in LDS
    __syncthreads();       // ...and is visible; h-tile writes visible too

    // prefetch next step's P tile into the other buffer
    if (s + 1 < Lsz) {
      const int tn = dir ? (Lsz - 2 - s) : (s + 1);
      const float* Pn = P + ((size_t)tn * Bsz + bg * 16) * Hsz;
#pragma unroll
      for (int j = 0; j < 4; ++j) {
        const int idx = tid + j * 256;
        async_copy_b128(Pn + idx * 4, &ptile[buf ^ 1][idx * 4]);
      }
    }

    // C init from LDS P tile
    const float* pt = &ptile[buf][0];
    v8f a0, a1;
#pragma unroll
    for (int r = 0; r < 8; ++r) {
      a0[r] = pt[(8 * hi + r) * Hsz + nt0 * 16 + col];
      a1[r] = pt[(8 * hi + r) * Hsz + nt1 * 16 + col];
    }

#pragma unroll
    for (int kc = 0; kc < 8; ++kc) {
      V16BF am, b0, b1;
      const unsigned short* ar = &hlds[col * Hsz + kc * 32 + hi * 8];
      am.q[0] = *(const u32x4*)ar;
      am.q[1] = *(const u32x4*)(ar + 16);
      const unsigned short* w0 =
          Whbf + (size_t)(nt0 * 16 + col) * Hsz + kc * 32 + hi * 8;
      const unsigned short* w1 =
          Whbf + (size_t)(nt1 * 16 + col) * Hsz + kc * 32 + hi * 8;
      b0.q[0] = *(const u32x4*)w0; b0.q[1] = *(const u32x4*)(w0 + 16);
      b1.q[0] = *(const u32x4*)w1; b1.q[1] = *(const u32x4*)(w1 + 16);
      a0 = __builtin_amdgcn_wmma_f32_16x16x32_bf16(
          false, am.v, false, b0.v, (short)0, a0, false, false);
      a1 = __builtin_amdgcn_wmma_f32_16x16x32_bf16(
          false, am.v, false, b1.v, (short)0, a1, false, false);
    }

    float hn0[8], hn1[8];
#pragma unroll
    for (int r = 0; r < 8; ++r) {
      float pre = a0[r];
      float sg = sigm(pre);
      cs0[r] = sg * (cs0[r] + tanhf(pre));
      hn0[r] = sg * tanhf(cs0[r]);
      pre = a1[r];
      sg = sigm(pre);
      cs1[r] = sg * (cs1[r] + tanhf(pre));
      hn1[r] = sg * tanhf(cs1[r]);
    }

    __syncthreads();   // everyone done reading this step's h tile + P tile
#pragma unroll
    for (int r = 0; r < 8; ++r) {
      const int m = 8 * hi + r;
      hlds[m * Hsz + nt0 * 16 + col] = f2bf(hn0[r]);
      hlds[m * Hsz + nt1 * 16 + col] = f2bf(hn1[r]);
      float* hb = hbuf + ((size_t)s * Bsz + bg * 16 + m) * Hsz;
      hb[nt0 * 16 + col] = hn0[r];
      hb[nt1 * 16 + col] = hn1[r];
      hl0[r] = hn0[r];
      hl1[r] = hn1[r];
    }
    buf ^= 1;
  }

  if (dir) {  // backward scan's final (h,c) seeds the attention scan
#pragma unroll
    for (int r = 0; r < 8; ++r) {
      const size_t row = (size_t)(bg * 16 + 8 * hi + r);
      hstate[row * Hsz + nt0 * 16 + col] = hl0[r];
      hstate[row * Hsz + nt1 * 16 + col] = hl1[r];
      cstate[row * Hsz + nt0 * 16 + col] = cs0[r];
      cstate[row * Hsz + nt1 * 16 + col] = cs1[r];
    }
  }
}

// enc = 0.5*(hfwd + hbwd), written in place over hfwd
__global__ void combine_kernel(float* __restrict__ encio,
                               const float* __restrict__ hb, size_t n) {
  size_t i = (size_t)blockIdx.x * blockDim.x + threadIdx.x;
  const size_t stride = (size_t)gridDim.x * blockDim.x;
  for (; i < n; i += stride) encio[i] = 0.5f * (encio[i] + hb[i]);
}

// ---------------------------------------------------------------------------
// global generation barrier (64 cooperating WGs)
// ---------------------------------------------------------------------------
__device__ __forceinline__ void gbarrier(int* bar, int nwg) {
  __syncthreads();
  if (threadIdx.x == 0) {
    volatile int* vb = bar;
    const int gen = vb[1];
    __threadfence();
    if (atomicAdd(&bar[0], 1) == nwg - 1) {
      bar[0] = 0;
      __threadfence();
      atomicAdd(&bar[1], 1);
    } else {
      while (vb[1] == gen) { __builtin_amdgcn_s_sleep(2); }
    }
    __threadfence();
  }
  __syncthreads();
}

// ---------------------------------------------------------------------------
// Attention scan. 64 WGs x 256 threads, cooperative.
// Per step: energies (enc . h, WG w owns l in [8w,8w+8)), redundant softmax,
// context (WG w owns batch row w), cell GEMM (wave0, tile mi=w>>4, ni=w&15,
// c-state register-resident across all 512 steps).
// ---------------------------------------------------------------------------
__global__ void attention_kernel(const float* __restrict__ enc,
                                 const float* __restrict__ P,
                                 const unsigned short* __restrict__ Whbf,
                                 float* __restrict__ hglob,
                                 const float* __restrict__ cstate,
                                 unsigned short* __restrict__ ctxbf,
                                 float* __restrict__ energies,
                                 float* __restrict__ listH,
                                 int* bar) {
  const int tid = threadIdx.x;
  const int wg  = blockIdx.x;
  const int lane = tid & 31;
  const int col  = lane & 15;
  const int hi   = lane >> 4;
  const int mi   = wg >> 4;   // batch tile
  const int ni   = wg & 15;   // hidden tile

  __shared__ float red[256];
  __shared__ float alpha[Lsz];

  float cs[8];
  if (tid < 32) {
#pragma unroll
    for (int r = 0; r < 8; ++r)
      cs[r] = cstate[(size_t)(mi * 16 + 8 * hi + r) * Hsz + ni * 16 + col];
  }

  for (int t = 0; t < Lsz; ++t) {
    // ---- 1) energies[l] = sum_{b,h} enc[l,b,h]*h[b,h]
    for (int li = 0; li < 8; ++li) {
      const int l = wg * 8 + li;
      const float* el = enc + (size_t)l * (Bsz * Hsz);
      float p = 0.f;
      for (int e = tid; e < Bsz * Hsz; e += 256) {
        __builtin_prefetch(el + e + 4096, 0, 1);   // global_prefetch_b8
        p += el[e] * hglob[e];
      }
      red[tid] = p; __syncthreads();
      for (int off = 128; off > 0; off >>= 1) {
        if (tid < off) red[tid] += red[tid + off];
        __syncthreads();
      }
      if (tid == 0) energies[l] = red[0];
      __syncthreads();
    }
    gbarrier(bar, 64);

    // ---- 2) softmax over 512 energies (redundant per WG)
    const float e0 = energies[tid];
    const float e1 = energies[tid + 256];
    red[tid] = fmaxf(e0, e1); __syncthreads();
    for (int off = 128; off > 0; off >>= 1) {
      if (tid < off) red[tid] = fmaxf(red[tid], red[tid + off]);
      __syncthreads();
    }
    const float M = red[0]; __syncthreads();
    const float x0 = __expf(e0 - M);
    const float x1 = __expf(e1 - M);
    red[tid] = x0 + x1; __syncthreads();
    for (int off = 128; off > 0; off >>= 1) {
      if (tid < off) red[tid] += red[tid + off];
      __syncthreads();
    }
    const float inv = 1.0f / red[0]; __syncthreads();
    alpha[tid] = x0 * inv;
    alpha[tid + 256] = x1 * inv;
    __syncthreads();

    // ---- 3) context[b,h] = h + sum_l alpha[l]*enc[l,b,h]; WG w -> batch w
    {
      const int e = wg * Hsz + tid;
      float acc = hglob[e];
      for (int l = 0; l < Lsz; ++l)
        acc += alpha[l] * enc[(size_t)l * (Bsz * Hsz) + e];
      ctxbf[e] = f2bf(acc);
    }
    gbarrier(bar, 64);

    // ---- 4) cell: pre = ctx @ Wh^T + P[t]; one WMMA tile, wave0 only
    if (tid < 32) {
      const float* Pt = P + (size_t)t * (Bsz * Hsz);
      v8f acc;
#pragma unroll
      for (int r = 0; r < 8; ++r)
        acc[r] = Pt[(size_t)(mi * 16 + 8 * hi + r) * Hsz + ni * 16 + col];

#pragma unroll
      for (int kc = 0; kc < 8; ++kc) {
        V16BF am, bm;
        const unsigned short* ar =
            ctxbf + (size_t)(mi * 16 + col) * Hsz + kc * 32 + hi * 8;
        am.q[0] = *(const u32x4*)ar;
        am.q[1] = *(const u32x4*)(ar + 16);
        const unsigned short* wr =
            Whbf + (size_t)(ni * 16 + col) * Hsz + kc * 32 + hi * 8;
        bm.q[0] = *(const u32x4*)wr;
        bm.q[1] = *(const u32x4*)(wr + 16);
        acc = __builtin_amdgcn_wmma_f32_16x16x32_bf16(
            false, am.v, false, bm.v, (short)0, acc, false, false);
      }

#pragma unroll
      for (int r = 0; r < 8; ++r) {
        const float pre = acc[r];
        const float sg = sigm(pre);
        cs[r] = sg * (cs[r] + tanhf(pre));
        const float hn = sg * tanhf(cs[r]);
        const size_t e = (size_t)(mi * 16 + 8 * hi + r) * Hsz + ni * 16 + col;
        hglob[e] = hn;
        listH[(size_t)t * (Bsz * Hsz) + e] = hn;
      }
    }
    gbarrier(bar, 64);
  }
}

// ---------------------------------------------------------------------------
// out[b*L+t, o] = (t > slen[b]) ? (0,1) : listH[t,b,:] . Wl[o,:] + bl[o]
// ---------------------------------------------------------------------------
__global__ void output_kernel(const float* __restrict__ listH,
                              const float* __restrict__ Wl,
                              const float* __restrict__ bl,
                              const int* __restrict__ slen,
                              float* __restrict__ out) {
  const int idx = blockIdx.x * blockDim.x + threadIdx.x;  // 0..32767
  const int b = idx / Lsz;
  const int t = idx % Lsz;
  if (t > slen[b]) {
    out[(size_t)idx * 2 + 0] = 0.0f;
    out[(size_t)idx * 2 + 1] = 1.0f;
    return;
  }
  const float* hr = listH + ((size_t)t * Bsz + b) * Hsz;
  float o0 = bl[0], o1 = bl[1];
  for (int n = 0; n < Hsz; ++n) {
    const float h = hr[n];
    o0 += h * Wl[n];
    o1 += h * Wl[Hsz + n];
  }
  out[(size_t)idx * 2 + 0] = o0;
  out[(size_t)idx * 2 + 1] = o1;
}

// ---------------------------------------------------------------------------
extern "C" void kernel_launch(void* const* d_in, const int* in_sizes, int n_in,
                              void* d_out, int out_size, void* d_ws, size_t ws_size,
                              hipStream_t stream) {
  const float* x    = (const float*)d_in[0];
  const int*   slen = (const int*)  d_in[1];
  const float* Wh   = (const float*)d_in[2];
  const float* bh   = (const float*)d_in[3];
  const float* Wx   = (const float*)d_in[4];
  const float* bx   = (const float*)d_in[5];
  const float* Wl   = (const float*)d_in[6];
  const float* bl   = (const float*)d_in[7];
  float* out = (float*)d_out;

  // ---- workspace carve-up (256B aligned) ----
  char* base = (char*)d_ws;
  size_t off = 0;
  auto carve = [&](size_t bytes) -> void* {
    void* p = base + off;
    off = (off + bytes + 255) & ~(size_t)255;
    return p;
  };
  const size_t LBH = (size_t)Lsz * Bsz * Hsz;       // 8,388,608 elems
  float*          P       = (float*)carve(LBH * 4); // x projection (+biases)
  float*          encbuf  = (float*)carve(LBH * 4); // hfwd, then enc
  float*          lbuf    = (float*)carve(LBH * 4); // hbwd, then listH
  unsigned short* Xbf     = (unsigned short*)carve(LBH * 2); // x in bf16, [l*B+b, k]
  unsigned short* Whbf    = (unsigned short*)carve((size_t)Hsz * Hsz * 2);
  unsigned short* Wxbf    = (unsigned short*)carve((size_t)Hsz * Dsz * 2);
  float*          hstate  = (float*)carve((size_t)Bsz * Hsz * 4);
  float*          cstate  = (float*)carve((size_t)Bsz * Hsz * 4);
  unsigned short* ctxbf   = (unsigned short*)carve((size_t)Bsz * Hsz * 2);
  float*          energies= (float*)carve(Lsz * 4);
  int*            bar     = (int*)carve(256);

  // barrier counters must be zero at every launch (ws is not re-poisoned)
  hipMemsetAsync(bar, 0, 256, stream);

  // conversions to bf16
  cvt_bf16_kernel<<<(Hsz * Hsz + 255) / 256, 256, 0, stream>>>(Wh, Whbf, Hsz * Hsz);
  cvt_bf16_kernel<<<(Hsz * Dsz + 255) / 256, 256, 0, stream>>>(Wx, Wxbf, Hsz * Dsz);
  cvt_x_kernel<<<(unsigned)(LBH / 256), 256, 0, stream>>>(x, Xbf);

  // P = x @ Wx^T + bx + bh : 32768x256x256 WMMA GEMM, 8 tiles per WG
  xproj_kernel<<<4096, 256, 0, stream>>>(Xbf, Wxbf, bx, bh, P);

  // fwd + bwd scans (8 persistent workgroups, async P-tile staging)
  scan_kernel<<<8, 256, 0, stream>>>(P, Whbf, encbuf, lbuf, hstate, cstate);

  // enc = 0.5*(hfwd + hbwd)
  combine_kernel<<<8192, 256, 0, stream>>>(encbuf, lbuf, LBH);

  // attention scan (cooperative, 64 WGs); listH overwrites hbwd buffer
  attention_kernel<<<64, 256, 0, stream>>>(encbuf, P, Whbf, hstate, cstate,
                                           ctxbf, energies, lbuf, bar);

  // final projection + length mask
  output_kernel<<<(Bsz * Lsz) / 256, 256, 0, stream>>>(lbuf, Wl, bl, slen, out);
}
